// MSAPairAttentionLayer_65377992179736
// MI455X (gfx1250) — matrix operand order repaired and checked by reading, not compiled, not run on hardware
//
#include <hip/hip_runtime.h>
#include <hip/hip_bf16.h>
#include <cmath>

// ---------------------------------------------------------------------------
// MSA Pair Attention layer for MI455X (gfx1250, wave32, WMMA + async-LDS)
// B=1, R=256, L=384, D=512, C=256, H=4, HD=128
// ---------------------------------------------------------------------------

#define LEN   384
#define DIM   512
#define NCH   256
#define HEADS 4
#define HDIM  128
#define NROWS (256 * 384)            // R*L = 98304
#define BK    32
#define ASTR  40                     // A-tile LDS stride (halves): 80B, 16B-aligned
#define BSTR  34                     // B-tile (transposed) LDS stride (halves)

typedef __bf16 bf16_t;
typedef __bf16 bf16x16 __attribute__((ext_vector_type(16)));
typedef float  f32x8   __attribute__((ext_vector_type(8)));
typedef int    v4i     __attribute__((ext_vector_type(4)));

// ---- CDNA5 async global->LDS path (guarded; falls back to VGPR staging) ----
#if defined(__has_builtin)
#if __has_builtin(__builtin_amdgcn_global_load_async_to_lds_b128) && \
    __has_builtin(__builtin_amdgcn_s_wait_asynccnt)
#define USE_ASYNC_LDS 1
#endif
#endif
#ifndef USE_ASYNC_LDS
#define USE_ASYNC_LDS 0
#endif

#if USE_ASYNC_LDS
__device__ __forceinline__ void async_copy_b128(const bf16_t* gsrc, bf16_t* ldst) {
  typedef __attribute__((address_space(1))) v4i as1_v4i;
  typedef __attribute__((address_space(3))) v4i as3_v4i;
  __builtin_amdgcn_global_load_async_to_lds_b128(
      (as1_v4i*)gsrc, (as3_v4i*)ldst, 0, 0);
}
#endif

// -------------------------- WMMA helpers -----------------------------------

__device__ __forceinline__ f32x8 wmma_bf16(bf16x16 a, bf16x16 b, f32x8 c) {
  // 8 args: (neg_a, A, neg_b, B, c_mod, C, reuse_a, reuse_b)
  return __builtin_amdgcn_wmma_f32_16x16x32_bf16(false, a, false, b, (short)0, c,
                                                 false, false);
}

// A fragment (16x32 bf16, row-major in LDS, stride ASTR halves).
// Lane layout (ISA 7.12.2): lanes 0-15 -> M=lane, K={0..7,16..23};
// lanes 16-31 -> M=lane-16, K={8..15,24..31}; 2 bf16 per VGPR.
__device__ __forceinline__ void load_a_frag(const bf16_t* lds, int lane, bf16x16& a) {
  const bf16_t* row = lds + (lane & 15) * ASTR;
  const int hs = lane >> 4;
#pragma unroll
  for (int v = 0; v < 8; ++v) {
    const int k = 2 * (v & 3) + 8 * hs + 16 * (v >> 2);
    a[2 * v]     = row[k];
    a[2 * v + 1] = row[k + 1];
  }
}

// B fragment (32x16 bf16) read from transposed LDS tile BsT[n][k], stride BSTR.
// Lane layout: lanes 0-15 -> N=lane, K={0..7,16..23}; lanes 16-31 -> N=lane-16, K={8..15,24..31}.
__device__ __forceinline__ void load_b_frag(const bf16_t* ldsT, int lane, bf16x16& b) {
  const bf16_t* col = ldsT + (lane & 15) * BSTR;
  const int hs = lane >> 4;
#pragma unroll
  for (int v = 0; v < 8; ++v) {
    const int k = 2 * (v & 3) + 8 * hs + 16 * (v >> 2);
    b[2 * v]     = col[k];
    b[2 * v + 1] = col[k + 1];
  }
}

// C/D tile store (f32 16x16, 8 VGPRs): lane n = lane&15, rows v + 8*(lane>>4).
__device__ __forceinline__ void store_bf16_tile(bf16_t* C, int ldc, f32x8 c, int lane) {
  const int n = lane & 15, hs = lane >> 4;
#pragma unroll
  for (int v = 0; v < 8; ++v)
    C[(long)(v + 8 * hs) * ldc + n] = (bf16_t)c[v];
}

__device__ __forceinline__ void store_f32_res_tile(float* C, const float* X, int ldc,
                                                   f32x8 c, int lane) {
  const int n = lane & 15, hs = lane >> 4;
#pragma unroll
  for (int v = 0; v < 8; ++v) {
    const long idx = (long)(v + 8 * hs) * ldc + n;
    C[idx] = X[idx] + c[v];
  }
}

// 8 WMMAs per wave on the current LDS buffers (wave tile = 32 rows x 64 cols)
__device__ __forceinline__ void compute_tiles(const bf16_t* As, const bf16_t* BsT,
                                              int wy, int wx, int lane, f32x8 acc[2][4]) {
  bf16x16 a0, a1, b[4];
  load_a_frag(As + (wy * 32) * ASTR,      lane, a0);
  load_a_frag(As + (wy * 32 + 16) * ASTR, lane, a1);
#pragma unroll
  for (int j = 0; j < 4; ++j)
    load_b_frag(BsT + (wx * 64 + j * 16) * BSTR, lane, b[j]);
#pragma unroll
  for (int j = 0; j < 4; ++j) {
    acc[0][j] = wmma_bf16(a0, b[j], acc[0][j]);
    acc[1][j] = wmma_bf16(a1, b[j], acc[1][j]);
  }
}

// --------------------- 128x128 block-tile GEMM core -------------------------
// 256 threads = 8 waves; each wave computes a 32x64 patch (2x4 WMMA tiles).
// Double-buffered LDS. A tile is staged memory->LDS with the CDNA5 async path
// (GLOBAL_LOAD_ASYNC_TO_LDS_B128, ASYNCcnt) when available; B tile goes through
// VGPRs because it is transposed on the way in. One barrier per K-step.

__device__ __forceinline__ void gemm_block_128x128(const bf16_t* __restrict__ Ablk, int lda,
                                                   const bf16_t* __restrict__ Bblk, int ldb,
                                                   int K, f32x8 acc[2][4]) {
  __shared__ bf16_t As[2][128 * ASTR];
  __shared__ bf16_t BsT[2][128 * BSTR];

  const int tid  = threadIdx.x;
  const int lane = tid & 31;
  const int wave = tid >> 5;
  const int wy   = wave >> 1;   // 0..3 : 32-row band
  const int wx   = wave & 1;    // 0..1 : 64-col band

#pragma unroll
  for (int i = 0; i < 2; ++i)
#pragma unroll
    for (int j = 0; j < 4; ++j)
      acc[i][j] = (f32x8)0.0f;

  // ---- stage tile 0 into buffer 0 ----
#pragma unroll
  for (int it = 0; it < 2; ++it) {
    const int q = tid + it * 256, row = q >> 2, cu = q & 3;       // A: 128x32
    const bf16_t* p = Ablk + (long)row * lda + cu * 8;
#if USE_ASYNC_LDS
    async_copy_b128(p, &As[0][row * ASTR + cu * 8]);
#else
    const uint4 v = *reinterpret_cast<const uint4*>(p);
    unsigned int* dst = reinterpret_cast<unsigned int*>(&As[0][row * ASTR + cu * 8]);
    dst[0] = v.x; dst[1] = v.y; dst[2] = v.z; dst[3] = v.w;
#endif
  }
#pragma unroll
  for (int it = 0; it < 2; ++it) {
    const int q = tid + it * 256, kk = q >> 4, cn = q & 15;       // B: 32x128
    const uint4 v = *reinterpret_cast<const uint4*>(Bblk + (long)kk * ldb + cn * 8);
    const bf16_t* e = reinterpret_cast<const bf16_t*>(&v);
#pragma unroll
    for (int i = 0; i < 8; ++i)
      BsT[0][(cn * 8 + i) * BSTR + kk] = e[i];
  }
#if USE_ASYNC_LDS
  __builtin_amdgcn_s_wait_asynccnt(0);
#endif
  __syncthreads();

  int cur = 0;
  for (int kt = 0; kt + BK < K; kt += BK) {
    const int nxt = cur ^ 1;

    // ---- stage next A tile: async memory->LDS (no VGPR round-trip) ----
#if USE_ASYNC_LDS
#pragma unroll
    for (int it = 0; it < 2; ++it) {
      const int q = tid + it * 256, row = q >> 2, cu = q & 3;
      async_copy_b128(Ablk + (long)row * lda + kt + BK + cu * 8,
                      &As[nxt][row * ASTR + cu * 8]);
    }
#else
    uint4 ra[2];
#pragma unroll
    for (int it = 0; it < 2; ++it) {
      const int q = tid + it * 256, row = q >> 2, cu = q & 3;
      const bf16_t* p = Ablk + (long)row * lda + kt + BK + cu * 8;
      ra[it] = *reinterpret_cast<const uint4*>(p);
      __builtin_prefetch(p + BK, 0, 3);
    }
#endif
    // ---- issue next B tile's global loads into registers ----
    uint4 rb[2];
#pragma unroll
    for (int it = 0; it < 2; ++it) {
      const int q = tid + it * 256, kk = q >> 4, cn = q & 15;
      const bf16_t* p = Bblk + (long)(kt + BK + kk) * ldb + cn * 8;
      rb[it] = *reinterpret_cast<const uint4*>(p);
      __builtin_prefetch(p + (long)BK * ldb, 0, 3);
    }

    // ---- 8 WMMAs on current buffer (hides async/global latency) ----
    compute_tiles(As[cur], BsT[cur], wy, wx, lane, acc);

    // ---- store staged registers into the other buffer ----
#if !USE_ASYNC_LDS
#pragma unroll
    for (int it = 0; it < 2; ++it) {
      const int q = tid + it * 256, row = q >> 2, cu = q & 3;
      unsigned int* dst = reinterpret_cast<unsigned int*>(&As[nxt][row * ASTR + cu * 8]);
      dst[0] = ra[it].x; dst[1] = ra[it].y; dst[2] = ra[it].z; dst[3] = ra[it].w;
    }
#endif
#pragma unroll
    for (int it = 0; it < 2; ++it) {
      const int q = tid + it * 256, kk = q >> 4, cn = q & 15;
      const bf16_t* e = reinterpret_cast<const bf16_t*>(&rb[it]);
#pragma unroll
      for (int i = 0; i < 8; ++i)
        BsT[nxt][(cn * 8 + i) * BSTR + kk] = e[i];
    }
#if USE_ASYNC_LDS
    __builtin_amdgcn_s_wait_asynccnt(0);
#endif
    __syncthreads();
    cur = nxt;
  }
  // ---- final tile ----
  compute_tiles(As[cur], BsT[cur], wy, wx, lane, acc);
}

// -------------------------- kernels ----------------------------------------

// Convert f32 weights -> bf16
__global__ void f32_to_bf16_kernel(const float* __restrict__ a, bf16_t* __restrict__ o, int n) {
  const int i = blockIdx.x * 256 + threadIdx.x;
  if (i < n) o[i] = (bf16_t)a[i];
}

// LayerNorm over D=512 of x [NROWS, 512]; one wave per row; write bf16.
__global__ void ln_msa_kernel(const float* __restrict__ x, const float* __restrict__ g,
                              const float* __restrict__ b, bf16_t* __restrict__ out) {
  const int lane = threadIdx.x & 31;
  const int wave = threadIdx.x >> 5;
  const long row = (long)blockIdx.x * 8 + wave;
  const float* xr = x + row * DIM;
  float vals[16];
  float s = 0.f, s2 = 0.f;
#pragma unroll
  for (int i = 0; i < 16; ++i) {
    const float v = xr[lane + i * 32];
    vals[i] = v; s += v; s2 += v * v;
  }
#pragma unroll
  for (int off = 16; off > 0; off >>= 1) {
    s  += __shfl_xor(s,  off);
    s2 += __shfl_xor(s2, off);
  }
  const float mu  = s * (1.0f / DIM);
  const float var = s2 * (1.0f / DIM) - mu * mu;
  const float inv = rsqrtf(var + 1e-5f);
#pragma unroll
  for (int i = 0; i < 16; ++i) {
    const int d = lane + i * 32;
    out[row * DIM + d] = (bf16_t)((vals[i] - mu) * inv * g[d] + b[d]);
  }
}

// Symmetrize pair_feat, LN over C=256, project to H=4 logits; one wave per (i,j).
__global__ void pair_logits_kernel(const float* __restrict__ pair, const float* __restrict__ g,
                                   const float* __restrict__ b, const float* __restrict__ Wp,
                                   const int* __restrict__ mask, float* __restrict__ logits,
                                   float scale) {
  const int lane = threadIdx.x & 31;
  const int wave = threadIdx.x >> 5;
  const int p = blockIdx.x * 8 + wave;   // 0 .. L*L-1
  const int i = p / LEN, j = p % LEN;
  const float* p1 = pair + ((long)i * LEN + j) * NCH;
  const float* p2 = pair + ((long)j * LEN + i) * NCH;
  float vals[8];
  float s = 0.f, s2 = 0.f;
#pragma unroll
  for (int t = 0; t < 8; ++t) {
    const int c = lane + t * 32;
    const float v = 0.5f * (p1[c] + p2[c]);
    vals[t] = v; s += v; s2 += v * v;
  }
#pragma unroll
  for (int off = 16; off > 0; off >>= 1) {
    s  += __shfl_xor(s,  off);
    s2 += __shfl_xor(s2, off);
  }
  const float mu  = s * (1.0f / NCH);
  const float var = s2 * (1.0f / NCH) - mu * mu;
  const float inv = rsqrtf(var + 1e-5f);
  float acc[HEADS] = {0.f, 0.f, 0.f, 0.f};
#pragma unroll
  for (int t = 0; t < 8; ++t) {
    const int c = lane + t * 32;
    const float v = ((vals[t] - mu) * inv * g[c] + b[c]) * scale;
#pragma unroll
    for (int h = 0; h < HEADS; ++h)
      acc[h] += v * Wp[c * HEADS + h];
  }
#pragma unroll
  for (int h = 0; h < HEADS; ++h)
#pragma unroll
    for (int off = 16; off > 0; off >>= 1)
      acc[h] += __shfl_xor(acc[h], off);
  if (lane == 0) {
    const bool dead = (mask[i] * mask[j]) == 0;
#pragma unroll
    for (int h = 0; h < HEADS; ++h)
      logits[(long)h * LEN * LEN + (long)i * LEN + j] = dead ? -3.4028235e38f : acc[h];
  }
}

// Softmax over rows of [H*L, L]; one wave per row; write bf16 attn.
__global__ void softmax_kernel(const float* __restrict__ logits, bf16_t* __restrict__ attn) {
  const int lane = threadIdx.x & 31;
  const int wave = threadIdx.x >> 5;
  const long row = (long)blockIdx.x * 8 + wave;   // 0 .. H*L-1
  const float* lr = logits + row * LEN;
  float v[12];
  float m = -3.4028235e38f;
#pragma unroll
  for (int t = 0; t < 12; ++t) {
    v[t] = lr[lane + t * 32];
    m = fmaxf(m, v[t]);
  }
#pragma unroll
  for (int off = 16; off > 0; off >>= 1)
    m = fmaxf(m, __shfl_xor(m, off));
  float s = 0.f;
#pragma unroll
  for (int t = 0; t < 12; ++t) {
    const float e = __expf(v[t] - m);
    v[t] = e; s += e;
  }
#pragma unroll
  for (int off = 16; off > 0; off >>= 1)
    s += __shfl_xor(s, off);
  const float inv = 1.0f / s;
#pragma unroll
  for (int t = 0; t < 12; ++t)
    attn[row * LEN + lane + t * 32] = (bf16_t)(v[t] * inv);
}

// GEMM [M,512] x [512,512]; Cf!=null -> f32 out with residual X, else bf16 out.
__global__ void wmma_gemm_proj(const bf16_t* __restrict__ A, const bf16_t* __restrict__ B,
                               bf16_t* __restrict__ Cbf, const float* __restrict__ Xres,
                               float* __restrict__ Cf) {
  const long m0 = (long)blockIdx.y * 128;
  const int  n0 = blockIdx.x * 128;
  f32x8 acc[2][4];
  gemm_block_128x128(A + m0 * DIM, DIM, B + n0, DIM, DIM, acc);
  const int lane = threadIdx.x & 31;
  const int wave = threadIdx.x >> 5;
  const int wy = wave >> 1, wx = wave & 1;
#pragma unroll
  for (int ti = 0; ti < 2; ++ti)
#pragma unroll
    for (int tj = 0; tj < 4; ++tj) {
      const long rbase = m0 + wy * 32 + ti * 16;
      const int  cbase = n0 + wx * 64 + tj * 16;
      if (Cf != nullptr)
        store_f32_res_tile(Cf + rbase * DIM + cbase, Xres + rbase * DIM + cbase, DIM,
                           acc[ti][tj], lane);
      else
        store_bf16_tile(Cbf + rbase * DIM + cbase, DIM, acc[ti][tj], lane);
    }
}

// Batched: out2[r,i,h*128+d] = sum_j attn[h,i,j] * V[r,j,h*128+d]
// blockIdx.x == head (N-tile of 128 == HD), blockIdx.y == 128-row band, z == r.
__global__ void wmma_gemm_attn(const bf16_t* __restrict__ attnM, const bf16_t* __restrict__ V,
                               bf16_t* __restrict__ out2) {
  const int  r  = blockIdx.z;              // 0..255
  const int  h  = blockIdx.x;              // 0..3
  const long m0 = (long)blockIdx.y * 128;  // 0..383
  const bf16_t* Ablk = attnM + (long)h * LEN * LEN + m0 * LEN;
  const bf16_t* Bblk = V + (long)r * LEN * DIM + h * HDIM;
  bf16_t*       Cblk = out2 + ((long)r * LEN + m0) * DIM + h * HDIM;
  f32x8 acc[2][4];
  gemm_block_128x128(Ablk, LEN, Bblk, DIM, LEN, acc);
  const int lane = threadIdx.x & 31;
  const int wave = threadIdx.x >> 5;
  const int wy = wave >> 1, wx = wave & 1;
#pragma unroll
  for (int ti = 0; ti < 2; ++ti)
#pragma unroll
    for (int tj = 0; tj < 4; ++tj)
      store_bf16_tile(Cblk + (long)(wy * 32 + ti * 16) * DIM + wx * 64 + tj * 16, DIM,
                      acc[ti][tj], lane);
}

// -------------------------- launcher ---------------------------------------

extern "C" void kernel_launch(void* const* d_in, const int* in_sizes, int n_in,
                              void* d_out, int out_size, void* d_ws, size_t ws_size,
                              hipStream_t stream) {
  (void)in_sizes; (void)n_in; (void)out_size; (void)ws_size;

  const float* x     = (const float*)d_in[0];   // [1,256,384,512]
  const float* pair  = (const float*)d_in[1];   // [1,384,384,256]
  const int*   mask  = (const int*)  d_in[2];   // [1,384]
  const float* gmsa  = (const float*)d_in[3];
  const float* bmsa  = (const float*)d_in[4];
  const float* gpair = (const float*)d_in[5];
  const float* bpair = (const float*)d_in[6];
  const float* Wpair = (const float*)d_in[7];   // [256,4]
  const float* Wv    = (const float*)d_in[8];   // [512,512]
  const float* Wout  = (const float*)d_in[9];   // [512,512]
  float* out = (float*)d_out;                   // [1,256,384,512]

  char* ws = (char*)d_ws;
  // workspace layout (bytes)
  bf16_t* msa    = (bf16_t*)(ws + 0);           // NROWS*512 bf16 (later reused as out2)
  bf16_t* V      = (bf16_t*)(ws + 100663296);   // NROWS*512 bf16
  float*  logits = (float*) (ws + 201326592);   // 4*384*384 f32
  bf16_t* attn   = (bf16_t*)(ws + 203685888);   // 4*384*384 bf16
  bf16_t* WvB    = (bf16_t*)(ws + 204865536);   // 512*512 bf16
  bf16_t* WoB    = (bf16_t*)(ws + 205389824);   // 512*512 bf16

  const float scale = 1.0f / sqrtf((float)(HDIM * LEN));  // HD^-0.5 / sqrt(L)

  // 1) weight conversion
  f32_to_bf16_kernel<<<1024, 256, 0, stream>>>(Wv,   WvB, DIM * DIM);
  f32_to_bf16_kernel<<<1024, 256, 0, stream>>>(Wout, WoB, DIM * DIM);
  // 2) LN(msa)
  ln_msa_kernel<<<NROWS / 8, 256, 0, stream>>>(x, gmsa, bmsa, msa);
  // 3) pair symmetrize + LN + per-head logits
  pair_logits_kernel<<<(LEN * LEN) / 8, 256, 0, stream>>>(pair, gpair, bpair, Wpair, mask,
                                                          logits, scale);
  // 4) softmax -> bf16 attn
  softmax_kernel<<<(HEADS * LEN) / 8, 256, 0, stream>>>(logits, attn);
  // 5) V = msa @ W_v   (bf16 out)
  wmma_gemm_proj<<<dim3(DIM / 128, NROWS / 128), 256, 0, stream>>>(msa, WvB, V, nullptr, nullptr);
  // 6) out2 = attn @ V  (reuses msa buffer)
  wmma_gemm_attn<<<dim3(HEADS, LEN / 128, 256), 256, 0, stream>>>(attn, V, msa);
  // 7) out = x + out2 @ W_out  (f32 + residual)
  wmma_gemm_proj<<<dim3(DIM / 128, NROWS / 128), 256, 0, stream>>>(msa, WoB, nullptr, x, out);
}
